// GNN_5119601017288
// MI455X (gfx1250) — matrix operand order repaired and checked by reading, not compiled, not run on hardware
//
#include <hip/hip_runtime.h>
#include <hip/hip_bf16.h>

typedef __bf16 bf16;
typedef __attribute__((ext_vector_type(16))) __bf16 v16bf;
typedef __attribute__((ext_vector_type(8)))  __bf16 bf16x8;
typedef __attribute__((ext_vector_type(8)))  float  v8f;
typedef int v4i_t __attribute__((vector_size(16)));   // GCC-style, matches builtin

#define N_NODES 20000
#define N_EDGES 40000
#define N_GRAPH 1000
#define ATOM_F  133
#define BOND_F  14
#define HDIM    64
#define DEDIM   128
#define NLAYERS 4
#define QCOLS   (129 * 64)   /* 8256: 128 h-channels + 1 bias row (b2 folded in) */
#define MPAD    20096        /* 20000 padded to multiple of 128 */
#define EPAD    40064        /* 40000 padded to multiple of 128 */
#define IN0PAD  160          /* 133 padded to multiple of 32 */

// ---------------------------------------------------------------------------
// Async global->LDS staging (CDNA5 GLOBAL_LOAD_ASYNC_TO_LDS_B128, ASYNCcnt)
// with compile-safe fallback to a synchronous copy.
// ---------------------------------------------------------------------------
#if defined(__has_builtin)
#if __has_builtin(__builtin_amdgcn_global_load_async_to_lds_b128)
#define HAVE_ASYNC_LDS 1
#endif
#endif
#ifndef HAVE_ASYNC_LDS
#define HAVE_ASYNC_LDS 0
#endif

__device__ __forceinline__ void stage16(const bf16* g, bf16* l) {
#if HAVE_ASYNC_LDS
    __builtin_amdgcn_global_load_async_to_lds_b128(
        (__attribute__((address_space(1))) v4i_t*)(uintptr_t)g,
        (__attribute__((address_space(3))) v4i_t*)(uintptr_t)l, 0, 0);
#else
    *(bf16x8*)l = *(const bf16x8*)g;
#endif
}

__device__ __forceinline__ void stage_wait() {
#if HAVE_ASYNC_LDS
#if __has_builtin(__builtin_amdgcn_s_wait_asynccnt)
    __builtin_amdgcn_s_wait_asynccnt(0);
#else
    asm volatile("s_wait_asynccnt 0x0" ::: "memory");
#endif
#endif
}

// ---------------------------------------------------------------------------
// Generic fp32 -> bf16 pack with zero padding:  dst[rp, cp] from src[r, c]
// ---------------------------------------------------------------------------
__global__ void pack_pad_bf16(const float* __restrict__ src, bf16* __restrict__ dst,
                              int rows, int cols, int rowsPad, int colsPad) {
    int idx = blockIdx.x * blockDim.x + threadIdx.x;
    int total = rowsPad * colsPad;
    if (idx >= total) return;
    int r = idx / colsPad, c = idx % colsPad;
    float v = (r < rows && c < cols) ? src[(size_t)r * cols + c] : 0.0f;
    dst[idx] = (bf16)v;
}

// ---------------------------------------------------------------------------
// W2 [128, in_dim*64] (+ b2 [in_dim*64]) -> W2p bf16 [in_pad, 8256]
// W2p[i, k*64+o] = W2[k, i*64+o] ; W2p[i, 128*64+o] = b2[i*64+o]
// ---------------------------------------------------------------------------
__global__ void permute_w2(const float* __restrict__ W2, const float* __restrict__ B2,
                           bf16* __restrict__ W2p, int in_dim, int in_pad) {
    int idx = blockIdx.x * blockDim.x + threadIdx.x;
    int total = in_pad * QCOLS;
    if (idx >= total) return;
    int i = idx / QCOLS, col = idx % QCOLS;
    int k = col >> 6, o = col & 63;
    float v = 0.0f;
    if (i < in_dim)
        v = (k < DEDIM) ? W2[(size_t)k * ((size_t)in_dim * 64) + (size_t)i * 64 + o]
                        : B2[(size_t)i * 64 + o];
    W2p[idx] = (bf16)v;
}

// ---------------------------------------------------------------------------
// WMMA fragment loaders from LDS (layouts per ISA §7.12.2).
// ---------------------------------------------------------------------------
__device__ __forceinline__ v16bf load_afrag(const bf16* p) {   // p = &As[m][kh]
    bf16x8 a0 = *(const bf16x8*)p;
    bf16x8 a1 = *(const bf16x8*)(p + 16);
    return __builtin_shufflevector(a0, a1, 0, 1, 2, 3, 4, 5, 6, 7,
                                   8, 9, 10, 11, 12, 13, 14, 15);
}
__device__ __forceinline__ v16bf load_bfrag(const bf16* p) {   // p = &Bs[n][kb]
    bf16x8 b0 = *(const bf16x8*)p;
    bf16x8 b1 = *(const bf16x8*)(p + 16);
    return __builtin_shufflevector(b0, b1, 0, 1, 2, 3, 4, 5, 6, 7,
                                   8, 9, 10, 11, 12, 13, 14, 15);
}

// ---------------------------------------------------------------------------
// Tiled bf16 WMMA GEMM: C[M,Nc] = A[M,K] @ B[K,Nc]  (+bias, optional relu)
// block = 256 threads = 8 waves (4 M x 2 N); each wave owns a 32x32 output
// (2x2 WMMA tiles -> 4 v_wmma per K-step, A/B fragments reused 2x each).
// Block tile 128(M) x 64(N); K step 32. A row-major (lda), B row-major (ldb).
// Output fp32 (Cf) or bf16 (Cb). M%128==0, K%32==0, Nc%64==0 (caller pads).
// ---------------------------------------------------------------------------
__global__ __launch_bounds__(256)
void gemm_bf16_wmma(const bf16* __restrict__ A, const bf16* __restrict__ B,
                    float* __restrict__ Cf, bf16* __restrict__ Cb,
                    const float* __restrict__ bias, int relu,
                    int M, int K, int Nc, int lda, int ldb) {
    __shared__ __align__(16) bf16 As[128][48];  // 96B row stride, 16B-aligned chunks
    __shared__ __align__(16) bf16 Bs[64][48];   // stored transposed: [n][k]

    const int t     = threadIdx.x;
    const int lane  = t & 31;
    const int wv    = t >> 5;       // 0..7
    const int waveM = wv & 3;       // 4 slabs of 32 rows
    const int waveN = wv >> 2;      // 2 slabs of 32 cols
    const int mBase = blockIdx.y * 128;
    const int nBase = blockIdx.x * 64;

    const int aRow = t >> 1;        // 0..127
    const int aCh  = (t & 1) * 16;  // 0 or 16
    const int bK   = t >> 3;        // 0..31
    const int bN   = (t & 7) * 8;   // 0..56

    v8f acc[2][2] = {};

    for (int k0 = 0; k0 < K; k0 += 32) {
        __syncthreads();
        // stage A tile 128x32 (two b128 per thread, async->LDS when available)
        const bf16* ga = A + (size_t)(mBase + aRow) * lda + k0 + aCh;
        stage16(ga, &As[aRow][aCh]);
        stage16(ga + 8, &As[aRow][aCh + 8]);
        // stage B tile 32(K)x64(N), transposed into LDS
        bf16x8 bv = *(const bf16x8*)(B + (size_t)(k0 + bK) * ldb + nBase + bN);
#pragma unroll
        for (int i = 0; i < 8; ++i) Bs[bN + i][bK] = bv[i];
        if (k0 + 32 < K) {
            __builtin_prefetch(ga + 32, 0, 1);
            __builtin_prefetch(B + (size_t)(k0 + 32 + bK) * ldb + nBase + bN, 0, 1);
        }
        stage_wait();
        __syncthreads();

        const int mr = waveM * 32 + (lane & 15);
        const int kh = (lane >> 4) * 8;
        const int nc = waveN * 32 + (lane & 15);
        const int kb = (lane >> 4) * 8;
        v16bf af[2], bfr[2];
        af[0]  = load_afrag(&As[mr][kh]);
        af[1]  = load_afrag(&As[mr + 16][kh]);
        bfr[0] = load_bfrag(&Bs[nc][kb]);
        bfr[1] = load_bfrag(&Bs[nc + 16][kb]);
#pragma unroll
        for (int i = 0; i < 2; ++i)
#pragma unroll
            for (int j = 0; j < 2; ++j)
                acc[i][j] = __builtin_amdgcn_wmma_f32_16x16x32_bf16(
                    false, af[i], false, bfr[j], (short)0, acc[i][j], false, false);
    }

    // epilogue: C/D layout -> VGPR r: lanes0-15 row=r, lanes16-31 row=8+r
#pragma unroll
    for (int i = 0; i < 2; ++i) {
#pragma unroll
        for (int j = 0; j < 2; ++j) {
            const int col  = nBase + waveN * 32 + j * 16 + (lane & 15);
            const int rowb = mBase + waveM * 32 + i * 16 + (lane >> 4) * 8;
            const float bv = bias ? bias[col] : 0.0f;
#pragma unroll
            for (int r = 0; r < 8; ++r) {
                float v = acc[i][j][r] + bv;
                if (relu) v = fmaxf(v, 0.0f);
                size_t idx = (size_t)(rowb + r) * Nc + col;
                if (Cb) Cb[idx] = (bf16)v;
                else    Cf[idx] = v;
            }
        }
    }
}

// ---------------------------------------------------------------------------
// Per-edge: msg[e,:] = [h[e,:] , 1] @ Q[src[e]] (129x64), atomically added to
// y[dst[e],:]. One wave per edge; lane owns output pair {2*lane, 2*lane+1}.
// ---------------------------------------------------------------------------
__global__ __launch_bounds__(256)
void edge_msg_scatter(const float* __restrict__ H1, const bf16* __restrict__ Q,
                      const int* __restrict__ eidx, float* __restrict__ y) {
    const int lane = threadIdx.x & 31;
    const int e = blockIdx.x * 8 + (threadIdx.x >> 5);
    if (e >= N_EDGES) return;
    const int src = eidx[e];
    const int dst = eidx[N_EDGES + e];
    const float* h = H1 + (size_t)e * DEDIM;
    const bf16*  q = Q + (size_t)src * QCOLS + 2 * lane;

    float acc0 = 0.0f, acc1 = 0.0f;
    for (int k0 = 0; k0 < DEDIM; k0 += 32) {
        float hv = h[k0 + lane];
#pragma unroll
        for (int j = 0; j < 32; ++j) {
            float hk = __shfl(hv, j, 32);
            union { unsigned u; bf16 b[2]; } p;
            p.u = *(const unsigned*)(q + (size_t)(k0 + j) * HDIM);
            acc0 += hk * (float)p.b[0];
            acc1 += hk * (float)p.b[1];
        }
    }
    {   // bias row (k = 128, weight 1.0)
        union { unsigned u; bf16 b[2]; } p;
        p.u = *(const unsigned*)(q + (size_t)DEDIM * HDIM);
        acc0 += (float)p.b[0];
        acc1 += (float)p.b[1];
    }
    unsafeAtomicAdd(&y[(size_t)dst * HDIM + 2 * lane], acc0);
    unsafeAtomicAdd(&y[(size_t)dst * HDIM + 2 * lane + 1], acc1);
}

// ---------------------------------------------------------------------------
// BatchNorm: per-channel mean / rstd over relu(y); one block per channel.
// ---------------------------------------------------------------------------
__global__ __launch_bounds__(256)
void bn_stats(const float* __restrict__ y, float* __restrict__ mean,
              float* __restrict__ rstd) {
    const int c = blockIdx.x, t = threadIdx.x;
    float s = 0.0f, s2 = 0.0f;
    for (int n = t; n < N_NODES; n += 256) {
        float v = fmaxf(y[(size_t)n * HDIM + c], 0.0f);
        s += v; s2 += v * v;
    }
    __shared__ float sh[256], sh2[256];
    sh[t] = s; sh2[t] = s2;
    __syncthreads();
    for (int off = 128; off > 0; off >>= 1) {
        if (t < off) { sh[t] += sh[t + off]; sh2[t] += sh2[t + off]; }
        __syncthreads();
    }
    if (t == 0) {
        float mu  = sh[0] / (float)N_NODES;
        float var = sh2[0] / (float)N_NODES - mu * mu;
        mean[c] = mu;
        rstd[c] = rsqrtf(var + 1e-5f);
    }
}

__global__ void bn_apply(const float* __restrict__ y, const float* __restrict__ mean,
                         const float* __restrict__ rstd, const float* __restrict__ gamma,
                         const float* __restrict__ beta, float* __restrict__ xo) {
    int idx = blockIdx.x * blockDim.x + threadIdx.x;
    if (idx >= N_NODES * HDIM) return;
    int c = idx & 63;
    float v = fmaxf(y[idx], 0.0f);
    xo[idx] = gamma[c] * (v - mean[c]) * rstd[c] + beta[c];
}

__global__ void zero_f32(float* __restrict__ p, int n) {
    int idx = blockIdx.x * blockDim.x + threadIdx.x;
    if (idx < n) p[idx] = 0.0f;
}

__global__ void pool_scatter(const float* __restrict__ x, const int* __restrict__ batch,
                             float* __restrict__ pooled, float* __restrict__ counts) {
    int idx = blockIdx.x * blockDim.x + threadIdx.x;
    if (idx >= N_NODES * HDIM) return;
    int n = idx >> 6, c = idx & 63;
    int g = batch[n];
    unsafeAtomicAdd(&pooled[(size_t)g * HDIM + c], x[idx]);
    if (c == 0) unsafeAtomicAdd(&counts[g], 1.0f);
}

__global__ void pool_div(const float* __restrict__ pooled, const float* __restrict__ counts,
                         float* __restrict__ out) {
    int idx = blockIdx.x * blockDim.x + threadIdx.x;
    if (idx >= N_GRAPH * HDIM) return;
    out[idx] = pooled[idx] / fmaxf(counts[idx >> 6], 1.0f);
}

// ---------------------------------------------------------------------------
extern "C" void kernel_launch(void* const* d_in, const int* in_sizes, int n_in,
                              void* d_out, int out_size, void* d_ws, size_t ws_size,
                              hipStream_t stream) {
    const float* x_in  = (const float*)d_in[0];
    const int*   eidx  = (const int*)d_in[1];
    const float* eattr = (const float*)d_in[2];
    const int*   batch = (const int*)d_in[3];
    const float* w1_0  = (const float*)d_in[4];
    const float* b1_0  = (const float*)d_in[5];
    const float* w2_0  = (const float*)d_in[6];
    const float* b2_0  = (const float*)d_in[7];
    const float* w1s   = (const float*)d_in[8];
    const float* b1s   = (const float*)d_in[9];
    const float* w2s   = (const float*)d_in[10];
    const float* b2s   = (const float*)d_in[11];
    const float* gamma = (const float*)d_in[12];
    const float* beta  = (const float*)d_in[13];
    (void)in_sizes; (void)n_in; (void)out_size; (void)ws_size;

    // ---- workspace bump allocator ----
    char* ws = (char*)d_ws;
    size_t off = 0;
    auto alloc = [&](size_t bytes) -> void* {
        void* p = ws + off;
        off = (off + bytes + 255) & ~(size_t)255;
        return p;
    };
    bf16*  AeBf  = (bf16*)alloc((size_t)EPAD * 32 * sizeof(bf16));
    bf16*  W1bf  = (bf16*)alloc((size_t)32 * DEDIM * sizeof(bf16));
    float* H1    = (float*)alloc((size_t)EPAD * DEDIM * sizeof(float));
    bf16*  Xbf   = (bf16*)alloc((size_t)MPAD * IN0PAD * sizeof(bf16));
    bf16*  W2p   = (bf16*)alloc((size_t)IN0PAD * QCOLS * sizeof(bf16));
    bf16*  Q     = (bf16*)alloc((size_t)MPAD * QCOLS * sizeof(bf16));
    float* ybuf  = (float*)alloc((size_t)N_NODES * HDIM * sizeof(float));
    float* xcur  = (float*)alloc((size_t)N_NODES * HDIM * sizeof(float));
    float* meanb = (float*)alloc(64 * sizeof(float));
    float* rstdb = (float*)alloc(64 * sizeof(float));
    float* pooled = (float*)alloc((size_t)N_GRAPH * HDIM * sizeof(float));
    float* counts = (float*)alloc((size_t)N_GRAPH * sizeof(float));

    // pack edge_attr once: [E,14] -> bf16 [EPAD,32]
    {
        int tot = EPAD * 32;
        pack_pad_bf16<<<(tot + 255) / 256, 256, 0, stream>>>(eattr, AeBf, N_EDGES, BOND_F,
                                                             EPAD, 32);
    }

    for (int i = 0; i < NLAYERS; ++i) {
        const int in_dim = (i == 0) ? ATOM_F : HDIM;
        const int in_pad = (i == 0) ? IN0PAD : HDIM;
        const float* W1 = (i == 0) ? w1_0 : w1s + (size_t)(i - 1) * BOND_F * DEDIM;
        const float* B1 = (i == 0) ? b1_0 : b1s + (size_t)(i - 1) * DEDIM;
        const float* W2 = (i == 0) ? w2_0 : w2s + (size_t)(i - 1) * DEDIM * HDIM * HDIM;
        const float* B2 = (i == 0) ? b2_0 : b2s + (size_t)(i - 1) * HDIM * HDIM;
        const float* Xs = (i == 0) ? x_in : xcur;

        // edge MLP stage 1: H1 = relu(Ae @ W1 + b1)   [EPAD,128]
        {
            int tot = 32 * DEDIM;
            pack_pad_bf16<<<(tot + 255) / 256, 256, 0, stream>>>(W1, W1bf, BOND_F, DEDIM,
                                                                 32, DEDIM);
            gemm_bf16_wmma<<<dim3(DEDIM / 64, EPAD / 128), 256, 0, stream>>>(
                AeBf, W1bf, H1, nullptr, B1, /*relu=*/1,
                EPAD, 32, DEDIM, /*lda=*/32, /*ldb=*/DEDIM);
        }
        // node-side contraction: Q = Xbf @ W2p   [MPAD, 8256] bf16
        {
            int tot = MPAD * in_pad;
            pack_pad_bf16<<<(tot + 255) / 256, 256, 0, stream>>>(Xs, Xbf, N_NODES, in_dim,
                                                                 MPAD, in_pad);
            int totw = in_pad * QCOLS;
            permute_w2<<<(totw + 255) / 256, 256, 0, stream>>>(W2, B2, W2p, in_dim, in_pad);
            gemm_bf16_wmma<<<dim3(QCOLS / 64, MPAD / 128), 256, 0, stream>>>(
                Xbf, W2p, nullptr, Q, nullptr, /*relu=*/0,
                MPAD, in_pad, QCOLS, /*lda=*/in_pad, /*ldb=*/QCOLS);
        }
        // gather/matvec/scatter + BN
        {
            int tot = N_NODES * HDIM;
            zero_f32<<<(tot + 255) / 256, 256, 0, stream>>>(ybuf, tot);
            edge_msg_scatter<<<(N_EDGES + 7) / 8, 256, 0, stream>>>(H1, Q, eidx, ybuf);
            bn_stats<<<HDIM, 256, 0, stream>>>(ybuf, meanb, rstdb);
            bn_apply<<<(tot + 255) / 256, 256, 0, stream>>>(ybuf, meanb, rstdb,
                                                            gamma + (size_t)i * HDIM,
                                                            beta + (size_t)i * HDIM, xcur);
        }
    }

    // graph mean pooling
    {
        int totp = N_GRAPH * HDIM;
        zero_f32<<<(totp + 255) / 256, 256, 0, stream>>>(pooled, totp);
        zero_f32<<<(N_GRAPH + 255) / 256, 256, 0, stream>>>(counts, N_GRAPH);
        int tot = N_NODES * HDIM;
        pool_scatter<<<(tot + 255) / 256, 256, 0, stream>>>(xcur, batch, pooled, counts);
        pool_div<<<(totp + 255) / 256, 256, 0, stream>>>(pooled, counts, (float*)d_out);
    }
}
